// Model_65678639890860
// MI455X (gfx1250) — matrix operand order repaired and checked by reading, not compiled, not run on hardware
//
#include <hip/hip_runtime.h>

// LSTM-style sequential scan, VEL=6, U=4, seqlen=4096.
// Latency-bound serial recurrence -> single wave32, whole scan in-kernel.
// ONE v_wmma_f32_16x16x32_f16 per step computes the complete pre-activation
// of all four gates:
//   - recurrent term via block-diagonal embedding (K = 4v+kk, 0..23)
//   - input term Wx_g[u,v]*x[v] via extra rows K = 24+v
//   - biases folded into the constant C accumulator operand
// Sigmoid/tanh use v_rcp_f32 (rcpf builtin) instead of IEEE division to keep
// the serial dependency chain short.

#define VELD 6
#define UD   4

typedef __attribute__((ext_vector_type(16))) _Float16 v16h;
typedef __attribute__((ext_vector_type(8)))  float    v8f;

__device__ __forceinline__ float sigmoidf_(float x) {
    return __builtin_amdgcn_rcpf(1.0f + __expf(-x));
}
__device__ __forceinline__ float tanhf_(float x) {
    float e = __expf(-2.0f * x);
    return (1.0f - e) * __builtin_amdgcn_rcpf(1.0f + e);
}

__global__ __launch_bounds__(32, 1)
void lstm_scan_wmma(const float* __restrict__ vel,
                    const float* __restrict__ h0,
                    const float* __restrict__ c0,
                    const float* __restrict__ Wix, const float* __restrict__ Wih, const float* __restrict__ bi,
                    const float* __restrict__ Wfx, const float* __restrict__ Wfh, const float* __restrict__ bf,
                    const float* __restrict__ Wox, const float* __restrict__ Woh, const float* __restrict__ bo,
                    const float* __restrict__ Wgx, const float* __restrict__ Wgh, const float* __restrict__ bg,
                    const float* __restrict__ linear, const float* __restrict__ bl,
                    const int* __restrict__ seqlen_p,
                    float* __restrict__ out)
{
    const int  lane = threadIdx.x;             // 0..31, wave32
    const bool hi   = lane >= 16;
    const int  N    = hi ? lane - 16 : lane;   // D/B column index
    const int  seqlen = *seqlen_p;

    __shared__ float sh_h[16 * UD];            // h matrix, rows >= VELD stay zero
    __shared__ float sh_x[16];                 // x / out vector, >= VELD stays zero
    __shared__ float sh_og[VELD * 2 * UD];     // staging for o / cbar gates

    // ---- zero + init LDS state -------------------------------------------
    for (int j = lane; j < 16 * UD; j += 32) sh_h[j] = 0.0f;
    if (lane < 16) sh_x[lane] = 0.0f;
    __syncthreads();
    if (lane < VELD) {
        #pragma unroll
        for (int u = 0; u < UD; ++u) sh_h[lane * UD + u] = h0[lane * UD + u];
        sh_x[lane] = vel[lane];
    }

    // ---- constant A matrix (f16), B mask, bias C --------------------------
    // A[M=(g,u)=4g+u][K]:
    //   K = 4v+kk (0..23)  -> Wh_g[(4v+u), kk]     (recurrent weights)
    //   K = 24+v  (24..29) -> Wx_g[u, v]           (input weights)
    //   K = 30,31          -> 0
    // 16-bit A layout: lane<16: M=lane, half j -> K = j (j<8) / j+8 (j>=8)
    //                  lane>=16: M=lane-16,  K = j+8 (j<8) / j+16 (j>=8)
    // 16-bit B layout: lane<16: N=lane,  half j -> K = j (j<8) / j+8 (j>=8)
    //                  lane>=16: N=lane-16,       K = j+8 (j<8) / j+16 (j>=8)
    const int gsel = N >> 2, uA = N & 3;
    const float* Wh = (gsel == 0) ? Wih : (gsel == 1) ? Wfh : (gsel == 2) ? Woh : Wgh;
    const float* Wx = (gsel == 0) ? Wix : (gsel == 1) ? Wfx : (gsel == 2) ? Wox : Wgx;

    v16h A;
    float bmask[16];
    #pragma unroll
    for (int j = 0; j < 16; ++j) {
        int Ka = hi ? ((j < 8) ? j + 8 : j + 16) : ((j < 8) ? j : j + 8);
        float aval = 0.0f;
        if (Ka < 24) {
            int v = Ka >> 2, kk = Ka & 3;
            aval = Wh[(v * UD + uA) * UD + kk];
        } else if (Ka < 24 + VELD) {
            int v = Ka - 24;
            aval = Wx[uA * VELD + v];
        }
        A[j] = (_Float16)aval;

        int Kb = hi ? ((j < 8) ? j + 8 : j + 16) : ((j < 8) ? j : j + 8);            // note: Kb & 3 == j & 3
        if (Kb < 24)  bmask[j] = ((Kb >> 2) == N) ? 1.0f : 0.0f;
        else          bmask[j] = ((Kb - 24) == N) ? 1.0f : 0.0f;  // x rows
    }

    // C = bias matrix: lane<16 holds M=r (g=0 rows 0..3 -> bi, g=1 rows 4..7 -> bf)
    //                  lane>=16 holds M=8+r (bo / bg)
    const float* bLo = hi ? bo : bi;
    const float* bHi = hi ? bg : bf;
    v8f cbias;
    #pragma unroll
    for (int r = 0; r < 8; ++r) {
        float bv = 0.0f;
        if (N < VELD) bv = (r < 4) ? bLo[N * UD + r] : bHi[N * UD + (r - 4)];
        cbias[r] = bv;
    }

    // ---- per-lane role constants -----------------------------------------
    const bool role1 = (lane < VELD);              // i,f gates + state update
    const bool role2 = (hi && N < VELD);           // o,cbar gates
    const int  v = N;
    float lin[UD] = {0,0,0,0}, cst[UD] = {0,0,0,0};
    float blv = 0.0f;
    if (role1) {
        #pragma unroll
        for (int u = 0; u < UD; ++u) {
            lin[u] = linear[v * UD + u];
            cst[u] = c0[v * UD + u];
        }
        blv = bl[v];
    }
    __syncthreads();

    // ---- sequential scan --------------------------------------------------
    for (int t = 0; t < seqlen; ++t) {
        // gather h row for this lane's column + current x
        float hrow[UD];
        #pragma unroll
        for (int kk = 0; kk < UD; ++kk) hrow[kk] = sh_h[N * UD + kk];
        float xv = sh_x[N];

        // rebuild block-diagonal B (f16): h rows for K<24, x rows for K>=24
        v16h B;
        #pragma unroll
        for (int j = 0; j < 16; ++j) {
            float val = (hi && j >= 8) ? xv : hrow[j & 3];
            B[j] = (_Float16)(val * bmask[j]);
        }

        // one WMMA -> full pre-activation of all 4 gates (incl. Wx*x + bias)
        v8f d = __builtin_amdgcn_wmma_f32_16x16x32_f16(
            /*neg_a=*/false, A, /*neg_b=*/false, B,
            /*c_mod=*/(short)0, cbias, /*reuse_a=*/false, /*reuse_b=*/false);

        // D layout: lane v (role1) holds M=0..7 (i: M=u, f: M=4+u)
        //           lane 16+v (role2) holds M=8..15 (o: M=8+u, cbar: M=12+u)
        float ga[UD], gb[UD];
        #pragma unroll
        for (int u = 0; u < UD; ++u) {
            ga[u] = sigmoidf_(d[u]);
            gb[u] = sigmoidf_(d[u + 4]);
        }
        if (role2) {
            #pragma unroll
            for (int u = 0; u < UD; ++u) {
                sh_og[v * 8 + u]     = ga[u];   // o gate
                sh_og[v * 8 + 4 + u] = gb[u];   // cbar
            }
        }
        __syncthreads();
        if (role1) {
            float s = 0.0f;
            #pragma unroll
            for (int u = 0; u < UD; ++u) {
                float ot = sh_og[v * 8 + u];
                float gt = sh_og[v * 8 + 4 + u];
                cst[u] = gb[u] * cst[u] + ga[u] * gt;        // f*c + i*cbar
                float hn = ot * sigmoidf_(cst[u]);           // o * sigmoid(c)
                s += lin[u] * hn;
                sh_h[v * UD + u] = hn;
            }
            sh_x[v] = tanhf_(s + blv);                       // next x / out
        }
        __syncthreads();
    }

    if (lane < VELD) out[lane] = sh_x[lane];
}

extern "C" void kernel_launch(void* const* d_in, const int* in_sizes, int n_in,
                              void* d_out, int out_size, void* d_ws, size_t ws_size,
                              hipStream_t stream) {
    (void)in_sizes; (void)n_in; (void)out_size; (void)d_ws; (void)ws_size;
    const float* vel    = (const float*)d_in[0];
    const float* h0     = (const float*)d_in[1];
    const float* c0     = (const float*)d_in[2];
    const float* Wix    = (const float*)d_in[3];
    const float* Wih    = (const float*)d_in[4];
    const float* bi     = (const float*)d_in[5];
    const float* Wfx    = (const float*)d_in[6];
    const float* Wfh    = (const float*)d_in[7];
    const float* bf     = (const float*)d_in[8];
    const float* Wox    = (const float*)d_in[9];
    const float* Woh    = (const float*)d_in[10];
    const float* bo     = (const float*)d_in[11];
    const float* Wgx    = (const float*)d_in[12];
    const float* Wgh    = (const float*)d_in[13];
    const float* bg     = (const float*)d_in[14];
    const float* linear = (const float*)d_in[15];
    const float* bl     = (const float*)d_in[16];
    const int*   seqlen = (const int*)d_in[17];
    float* out = (float*)d_out;

    lstm_scan_wmma<<<1, 32, 0, stream>>>(vel, h0, c0,
                                         Wix, Wih, bi,
                                         Wfx, Wfh, bf,
                                         Wox, Woh, bo,
                                         Wgx, Wgh, bg,
                                         linear, bl, seqlen, out);
}
